// GCN_Class_82514911691357
// MI455X (gfx1250) — matrix operand order repaired and checked by reading, not compiled, not run on hardware
//
#include <hip/hip_runtime.h>
#include <hip/hip_bf16.h>
#include <math.h>

// ---------------------------------------------------------------------------
// GCN forward for MI455X (gfx1250, wave32, WMMA).
//   h1 = segment_sum(w * (x@W1)[src] -> dst) + b1
//   h2 = segment_sum(w * (h1@W2)[src] -> dst) + b2
//   out = log_softmax(h2)
// Dense transforms use v_wmma_f32_16x16x32_f16 with an f16 hi/lo split
// (3 WMMAs per tile) for ~f32 accuracy. Edge phase is the memory-bound part;
// node tables (25.6 MB) are L2-resident on the 192 MB L2.
// ---------------------------------------------------------------------------

typedef __attribute__((ext_vector_type(16))) _Float16 v16h;
typedef __attribute__((ext_vector_type(8)))  float    v8f;

// ---------------------------------------------------------------------------
// Pack a row-major f32 weight matrix W[K][N] into the wave32 WMMA B-operand
// register layout (contiguous k, matching the A fragments in wmma_gemm):
//   lane L holds column n = nt*16 + (L&15)
//   halves h=0..15 hold K = ki*32 + ((L>=16)?16:0) + h
// Produces hi/lo f16 split for near-f32 precision. Columns >= N are zero pad.
// ---------------------------------------------------------------------------
__global__ void pack_w_kernel(const float* __restrict__ W,
                              _Float16* __restrict__ hi,
                              _Float16* __restrict__ lo,
                              int K, int N, int KI, int NT)
{
    int idx   = blockIdx.x * blockDim.x + threadIdx.x;
    int total = NT * KI * 32 * 16;
    if (idx >= total) return;
    int h    = idx & 15;
    int lane = (idx >> 4) & 31;
    int t    = idx >> 9;            // nt*KI + ki
    int ki   = t % KI;
    int nt   = t / KI;
    int n    = nt * 16 + (lane & 15);
    int k    = ki * 32 + ((lane & 16) ? 16 : 0) + h;
    float v  = (n < N) ? W[(size_t)k * N + n] : 0.0f;
    _Float16 vh = (_Float16)v;
    hi[idx] = vh;
    lo[idx] = (_Float16)(v - (float)vh);
}

// ---------------------------------------------------------------------------
// WMMA GEMM: out[M x (NT*16)] = A[M x K] @ Bpacked, f32 in/out.
// One wave computes a 16-row stripe across all NT 16-col tiles.
// A is converted f32 -> f16 hi/lo on the fly, loaded with contiguous k in the A-operand
// layout (16-bit A 16x32: pairs per VGPR, K 0-15/16-31 split across lane halves).
// ---------------------------------------------------------------------------
template <int KI, int NT>
__global__ __launch_bounds__(256) void wmma_gemm(const float* __restrict__ A,
                                                 const _Float16* __restrict__ Bhi,
                                                 const _Float16* __restrict__ Blo,
                                                 float* __restrict__ out,
                                                 int mtiles, int nrows)
{
    const int K     = KI * 32;
    const int ldOut = NT * 16;
    int lane = threadIdx.x & 31;
    int wid  = threadIdx.x >> 5;
    int mt   = blockIdx.x * (blockDim.x >> 5) + wid;
    if (mt >= mtiles) return;              // wave-uniform: EXEC stays all-ones
    int m0    = mt * 16;
    int row   = m0 + (lane & 15);
    if (row >= nrows) row = nrows - 1;     // clamp (tail tile safety)
    int kbase = (lane & 16) ? 16 : 0;
    const float* arow = A + (size_t)row * K;

    v8f c[NT] = {};

    for (int ki = 0; ki < KI; ++ki) {
        // ---- build A hi/lo fragments (k = 16*half + s, matching packed B) ----
        v16h ahi, alo;
#pragma unroll
        for (int i = 0; i < 8; ++i) {
            int k = ki * 32 + kbase + (i << 1);
            float2 p = *(const float2*)(arow + k);
            _Float16 h0 = (_Float16)p.x;
            _Float16 h1 = (_Float16)p.y;
            ahi[2 * i]     = h0;
            ahi[2 * i + 1] = h1;
            alo[2 * i]     = (_Float16)(p.x - (float)h0);
            alo[2 * i + 1] = (_Float16)(p.y - (float)h1);
        }
        // ---- NT column tiles: 3 WMMAs each (AhBh + AlBh + AhBl) ----
#pragma unroll
        for (int nt = 0; nt < NT; ++nt) {
            const v16h* bp = (const v16h*)Bhi + ((size_t)(nt * KI + ki) * 32 + lane);
            const v16h* bq = (const v16h*)Blo + ((size_t)(nt * KI + ki) * 32 + lane);
            v16h bhi = *bp;
            v16h blo = *bq;
            c[nt] = __builtin_amdgcn_wmma_f32_16x16x32_f16(false, ahi, false, bhi,
                                                           (short)0, c[nt], false, false);
            c[nt] = __builtin_amdgcn_wmma_f32_16x16x32_f16(false, alo, false, bhi,
                                                           (short)0, c[nt], false, false);
            c[nt] = __builtin_amdgcn_wmma_f32_16x16x32_f16(false, ahi, false, blo,
                                                           (short)0, c[nt], false, false);
        }
    }

    // ---- store D (VGPR v: lanes 0-15 -> M=v, lanes 16-31 -> M=v+8) ----
    int rbase = m0 + ((lane & 16) ? 8 : 0);
#pragma unroll
    for (int nt = 0; nt < NT; ++nt) {
        int col = nt * 16 + (lane & 15);
#pragma unroll
        for (int v = 0; v < 8; ++v) {
            int r = rbase + v;
            if (r < nrows)
                out[(size_t)r * ldOut + col] = c[nt][v];
        }
    }
}

// ---------------------------------------------------------------------------
// agg[node][f] = bias[f]  (bias folded into the accumulator init)
// ---------------------------------------------------------------------------
__global__ void init_bias_kernel(float* __restrict__ agg,
                                 const float* __restrict__ b,
                                 size_t total, int D)
{
    size_t i = (size_t)blockIdx.x * blockDim.x + threadIdx.x;
    if (i < total) agg[i] = b[i % D];
}

// ---------------------------------------------------------------------------
// Edge scatter: one wave per edge.  agg[dst][f] += w * support[src][f].
// Coalesced 128 B gathers per wave iteration; FP32 HW atomics for the scatter.
// ---------------------------------------------------------------------------
__global__ __launch_bounds__(256) void edge_aggregate(const long long* __restrict__ src,
                                                      const long long* __restrict__ dst,
                                                      const float* __restrict__ ew,
                                                      const float* __restrict__ support,
                                                      float* __restrict__ agg,
                                                      int nE, int sStride, int aStride, int D)
{
    int wave = blockIdx.x * (blockDim.x >> 5) + (threadIdx.x >> 5);
    int lane = threadIdx.x & 31;
    if (wave >= nE) return;
    long long s = src[wave];
    long long d = dst[wave];
    float w = ew[wave];
    const float* sp = support + (size_t)s * sStride;
    float*       ap = agg     + (size_t)d * aStride;
    for (int f = lane; f < D; f += 32)
        unsafeAtomicAdd(ap + f, w * sp[f]);   // -> global_atomic_add_f32
}

// ---------------------------------------------------------------------------
// Row-wise log_softmax, one wave (32 lanes) per row of D<=64 values.
// ---------------------------------------------------------------------------
__global__ __launch_bounds__(256) void log_softmax_rows(const float* __restrict__ agg,
                                                        float* __restrict__ out,
                                                        int nRows, int D)
{
    int wave = blockIdx.x * (blockDim.x >> 5) + (threadIdx.x >> 5);
    int lane = threadIdx.x & 31;
    if (wave >= nRows) return;
    const float* r = agg + (size_t)wave * D;
    float v0 = (lane      < D) ? r[lane]      : -INFINITY;
    float v1 = (lane + 32 < D) ? r[lane + 32] : -INFINITY;
    float m = fmaxf(v0, v1);
#pragma unroll
    for (int off = 16; off > 0; off >>= 1)
        m = fmaxf(m, __shfl_xor(m, off, 32));
    float s = 0.0f;
    if (lane      < D) s += expf(v0 - m);
    if (lane + 32 < D) s += expf(v1 - m);
#pragma unroll
    for (int off = 16; off > 0; off >>= 1)
        s += __shfl_xor(s, off, 32);
    float ls = m + logf(s);
    float* o = out + (size_t)wave * D;
    if (lane      < D) o[lane]      = v0 - ls;
    if (lane + 32 < D) o[lane + 32] = v1 - ls;
}

// ---------------------------------------------------------------------------
// Host-side launch sequence (graph-capture safe: only kernel launches).
// ---------------------------------------------------------------------------
extern "C" void kernel_launch(void* const* d_in, const int* in_sizes, int n_in,
                              void* d_out, int out_size, void* d_ws, size_t ws_size,
                              hipStream_t stream)
{
    const int D_IN = 256, D_HID = 64, N_CLASS = 40;
    const float*     x    = (const float*)d_in[0];
    const long long* esrc = (const long long*)d_in[1];  // reference dtype: int64
    const long long* edst = (const long long*)d_in[2];
    const float*     ew   = (const float*)d_in[3];
    const float*     W1   = (const float*)d_in[4];
    const float*     b1   = (const float*)d_in[5];
    const float*     W2   = (const float*)d_in[6];
    const float*     b2   = (const float*)d_in[7];
    float*           out  = (float*)d_out;

    const int nNodes = in_sizes[0] / D_IN;     // 100000
    const int nEdges = in_sizes[1];            // 1600000

    // GEMM geometry
    const int KI1 = D_IN / 32;                 // 8
    const int NT1 = D_HID / 16;                // 4
    const int KI2 = D_HID / 32;                // 2
    const int NT2 = (N_CLASS + 15) / 16;       // 3 (40 -> padded 48)
    const int N2P = NT2 * 16;                  // 48

    // ---- carve workspace (256 B aligned) ----
    char* p = (char*)d_ws;
    auto carve = [&](size_t bytes) -> void* {
        void* r = (void*)p;
        p += (bytes + 255) & ~(size_t)255;
        return r;
    };
    float*    support1 = (float*)carve((size_t)nNodes * D_HID * sizeof(float));
    float*    agg1     = (float*)carve((size_t)nNodes * D_HID * sizeof(float));
    float*    support2 = (float*)carve((size_t)nNodes * N2P  * sizeof(float));
    _Float16* w1hi     = (_Float16*)carve((size_t)NT1 * KI1 * 512 * sizeof(_Float16));
    _Float16* w1lo     = (_Float16*)carve((size_t)NT1 * KI1 * 512 * sizeof(_Float16));
    _Float16* w2hi     = (_Float16*)carve((size_t)NT2 * KI2 * 512 * sizeof(_Float16));
    _Float16* w2lo     = (_Float16*)carve((size_t)NT2 * KI2 * 512 * sizeof(_Float16));
    float*    agg2     = support1;             // alias: support1 dead after layer-1 scatter
    (void)ws_size; (void)n_in; (void)out_size;

    const int mtiles = (nNodes + 15) / 16;     // 6250
    const int wavesPerBlock = 256 / 32;        // 8

    // 1) pack weights into WMMA B-operand layout (hi/lo split)
    {
        int total1 = NT1 * KI1 * 512;
        pack_w_kernel<<<(total1 + 255) / 256, 256, 0, stream>>>(W1, w1hi, w1lo, D_IN, D_HID, KI1, NT1);
        int total2 = NT2 * KI2 * 512;
        pack_w_kernel<<<(total2 + 255) / 256, 256, 0, stream>>>(W2, w2hi, w2lo, D_HID, N_CLASS, KI2, NT2);
    }

    // 2) support1 = x @ W1   (WMMA, f16 split, f32 accumulate)
    wmma_gemm<8, 4><<<(mtiles + wavesPerBlock - 1) / wavesPerBlock, 256, 0, stream>>>(
        x, w1hi, w1lo, support1, mtiles, nNodes);

    // 3) agg1 = b1 (broadcast)
    {
        size_t total = (size_t)nNodes * D_HID;
        init_bias_kernel<<<(unsigned)((total + 255) / 256), 256, 0, stream>>>(agg1, b1, total, D_HID);
    }

    // 4) agg1[dst] += w * support1[src]
    edge_aggregate<<<(nEdges + wavesPerBlock - 1) / wavesPerBlock, 256, 0, stream>>>(
        esrc, edst, ew, support1, agg1, nEdges, D_HID, D_HID, D_HID);

    // 5) support2 = agg1 @ W2  (padded to 48 cols; pad cols are exactly 0)
    wmma_gemm<2, 3><<<(mtiles + wavesPerBlock - 1) / wavesPerBlock, 256, 0, stream>>>(
        agg1, w2hi, w2lo, support2, mtiles, nNodes);

    // 6) agg2 = b2 (broadcast)
    {
        size_t total = (size_t)nNodes * N_CLASS;
        init_bias_kernel<<<(unsigned)((total + 255) / 256), 256, 0, stream>>>(agg2, b2, total, N_CLASS);
    }

    // 7) agg2[dst] += w * support2[src]   (read stride 48, write stride 40)
    edge_aggregate<<<(nEdges + wavesPerBlock - 1) / wavesPerBlock, 256, 0, stream>>>(
        esrc, edst, ew, support2, agg2, nEdges, N2P, N_CLASS, N_CLASS);

    // 8) out = log_softmax(agg2, axis=1)
    log_softmax_rows<<<(nNodes + wavesPerBlock - 1) / wavesPerBlock, 256, 0, stream>>>(
        agg2, out, nNodes, N_CLASS);
}
